// PhonemeLevelEncoder_66898410602539
// MI455X (gfx1250) — matrix operand order repaired and checked by reading, not compiled, not run on hardware
//
#include <hip/hip_runtime.h>
#include <hip/hip_bf16.h>

#define NB   32
#define CIN  192
#define TT   8192
#define HH   192
#define NPH  1024
#define NOUT 4
#define TPAD (NPH + 2)   // activation buffers padded with zero rows at t=-1 and t=NPH

typedef __attribute__((ext_vector_type(16))) __bf16 v16bf;
typedef __attribute__((ext_vector_type(8)))  __bf16 v8bf;
typedef __attribute__((ext_vector_type(8)))  float  v8f;

// Build a 16-elem bf16 fragment from two 8-elem (16-byte) chunks.
__device__ __forceinline__ v16bf load_frag16(const __bf16* p0, const __bf16* p1) {
    v8bf lo = *(const v8bf*)p0;
    v8bf hi = *(const v8bf*)p1;
    v16bf f;
#pragma unroll
    for (int j = 0; j < 8; ++j) { f[j] = lo[j]; f[j + 8] = hi[j]; }
    return f;
}

// A fragment (16x32 bf16), row-major weights:
// lane L<16 -> row M=L, K={0..7,16..23}; lane>=16 -> row M=L-16, K += 8.
__device__ __forceinline__ v16bf load_a(const __bf16* Wrow0, int K, int rowBase, int kBase, int lane) {
    int row  = rowBase + (lane & 15);
    int koff = (lane >> 4) << 3;                  // 0 or 8
    const __bf16* p = Wrow0 + (size_t)row * K + kBase + koff;
    return load_frag16(p, p + 16);
}

// ---------------- duration inclusive scan (per batch) ----------------
__global__ void scan_kernel(const int* __restrict__ w, int* __restrict__ ends) {
    __shared__ int s[NPH];
    int b = blockIdx.x, t = threadIdx.x;
    s[t] = w[b * NPH + t];
    __syncthreads();
    for (int off = 1; off < NPH; off <<= 1) {
        int v = (t >= off) ? s[t - off] : 0;
        __syncthreads();
        s[t] += v;
        __syncthreads();
    }
    ends[b * NPH + t] = s[t];
}

// ---------------- weight repacks to bf16 ----------------
__global__ void cvt_mat_kernel(const float* __restrict__ src, __bf16* __restrict__ dst, int n) {
    int i = blockIdx.x * blockDim.x + threadIdx.x;
    if (i < n) dst[i] = (__bf16)src[i];
}

// conv weights [O][I][3] -> [tap][O][I] so each A row is contiguous
__global__ void cvt_conv_kernel(const float* __restrict__ src, __bf16* __restrict__ dst) {
    int i = blockIdx.x * blockDim.x + threadIdx.x;
    if (i < HH * HH * 3) {
        int tap = i % 3, rem = i / 3;
        int ii = rem % HH, o = rem / HH;
        dst[((size_t)tap * HH + o) * HH + ii] = (__bf16)src[i];
    }
}

// ---------------- zero the t=-1 / t=NPH halo rows of activation buffers ----------------
__global__ void zero_halo_kernel(__bf16* __restrict__ a, __bf16* __restrict__ bb,
                                 __bf16* __restrict__ c) {
    int i = blockIdx.x * blockDim.x + threadIdx.x;   // NB * 2 * HH
    if (i >= NB * 2 * HH) return;
    int cc = i % HH, r = (i / HH) & 1, b = i / (2 * HH);
    size_t off = (size_t)b * TPAD * HH + (size_t)(r ? (NPH + 1) : 0) * HH + cc;
    a[off] = (__bf16)0.f;
    bb[off] = (__bf16)0.f;
    c[off] = (__bf16)0.f;
}

// ---------------- segment mean (x read once, t-contiguous) ----------------
// Output t-major: spec[b][ph+1][c]
__global__ void segmean_kernel(const float* __restrict__ x, const int* __restrict__ ends,
                               __bf16* __restrict__ spec) {
    int bc = blockIdx.x;               // b*CIN + c
    int b = bc / CIN, c = bc % CIN;
    const float* xr = x + ((size_t)b * CIN + c) * TT;
    const int*   e  = ends + b * NPH;
    __bf16* sp = spec + (size_t)b * TPAD * HH + c;
    for (int ph = threadIdx.x; ph < NPH; ph += blockDim.x) {
        int e1 = e[ph];
        int e0 = ph ? e[ph - 1] : 0;
        int n  = e1 - e0;
        float s = 0.f;
        for (int t = e0; t < e1; ++t) s += xr[t];
        float v = (n > 0) ? s / (float)n : 0.f;
        sp[(size_t)(ph + 1) * HH] = (__bf16)v;
    }
}

// ---------------- WMMA GEMM / conv(+ReLU+LN) block, t-major activations ----------------
// Y[b][t] = sum_tap W_tap (192x192) @ in[b][t+tap-HALO] + bias, optional ReLU+LayerNorm.
// Activations [b][TPAD][HH] (t-major): B fragment = per-lane two 8-bf16 chunks (k, k+16) from L2.
// Grid: (NPH/64, NB). Block: 256 = 8 waves; each wave owns 3 M-tiles x 2 N-tiles.
template <int TAPS, bool DO_LN>
__global__ __launch_bounds__(256) void block_kernel(
    const __bf16* __restrict__ W,     // [TAPS][192][192]
    const float*  __restrict__ bias,  // [192]
    const __bf16* __restrict__ in,    // [NB][TPAD][HH], halo rows zero
    const float*  __restrict__ mask,  // [NB][NPH]
    const float*  __restrict__ lng,   // LN gamma or null
    const float*  __restrict__ lnb,   // LN beta or null
    __bf16*       __restrict__ out)   // [NB][TPAD][HH]
{
    constexpr int HALO = TAPS / 2;
    __shared__ float sy[DO_LN ? (64 * HH) : 1];
    __shared__ float sp1[DO_LN ? 256 : 1];
    __shared__ float sp2[DO_LN ? 256 : 1];
    __shared__ float smean[DO_LN ? 64 : 1];
    __shared__ float srstd[DO_LN ? 64 : 1];

    const int b  = blockIdx.y;
    const int t0 = blockIdx.x * 64;
    const int tid = threadIdx.x;
    const int lane = tid & 31, wid = tid >> 5;
    const int wave_m = wid >> 1;      // 0..3 -> M tiles [wave_m*48, +48)
    const int wave_n = wid & 1;       // 0..1 -> N tiles [wave_n*32, +32)
    const int nlane = lane & 15;
    const int koffB = (lane >> 4) << 3;   // B layout: lanes>=16 hold K+8 (K={0..7,16..23}+koffB)
    const int moff  = (lane >> 4) << 3;   // C/D layout: lanes>=16 hold M+8

    const __bf16* src = in + (size_t)b * TPAD * HH;

    // Prefetch next t-tile's activation rows into L2 (global_prefetch_b8).
    {
        int pr = t0 + 64 + (tid & 63);
        if (pr < NPH) __builtin_prefetch(src + (size_t)(pr + 1) * HH, 0, 0);
    }

    v8f acc[3][2] = {};

#pragma unroll
    for (int kt = 0; kt < 6; ++kt) {              // K = 192 = 6 * 32
        const int kBase = kt * 32;
        v16bf bfr[2][TAPS];
#pragma unroll
        for (int nt = 0; nt < 2; ++nt)
#pragma unroll
            for (int tap = 0; tap < TAPS; ++tap) {
                int row = t0 + wave_n * 32 + nt * 16 + nlane + tap - HALO + 1;
                const __bf16* p = src + (size_t)row * HH + kBase + koffB;
                bfr[nt][tap] = load_frag16(p, p + 16);
            }
#pragma unroll
        for (int mt = 0; mt < 3; ++mt) {
#pragma unroll
            for (int tap = 0; tap < TAPS; ++tap) {
                v16bf a = load_a(W + (size_t)tap * HH * HH, HH,
                                 wave_m * 48 + mt * 16, kBase, lane);
#pragma unroll
                for (int nt = 0; nt < 2; ++nt)
                    acc[mt][nt] = __builtin_amdgcn_wmma_f32_16x16x32_bf16(
                        false, a, false, bfr[nt][tap], (short)0, acc[mt][nt], false, false);
            }
        }
    }

    __bf16* dst = out + (size_t)b * TPAD * HH;

    if constexpr (!DO_LN) {
        // epilogue: (acc + bias) * mask -> bf16, one 16B store per tile per lane
#pragma unroll
        for (int mt = 0; mt < 3; ++mt)
#pragma unroll
            for (int nt = 0; nt < 2; ++nt) {
                int m0 = wave_m * 48 + mt * 16 + moff;
                int nn = wave_n * 32 + nt * 16 + nlane;
                float mk = mask[b * NPH + t0 + nn];
                v8bf o;
#pragma unroll
                for (int v = 0; v < 8; ++v)
                    o[v] = (__bf16)((acc[mt][nt][v] + bias[m0 + v]) * mk);
                *(v8bf*)&dst[(size_t)(t0 + nn + 1) * HH + m0] = o;
            }
    } else {
        // epilogue: relu((acc+bias)*mask) -> LDS [col][192] -> LayerNorm -> *mask -> bf16
#pragma unroll
        for (int mt = 0; mt < 3; ++mt)
#pragma unroll
            for (int nt = 0; nt < 2; ++nt) {
                int m0 = wave_m * 48 + mt * 16 + moff;
                int nn = wave_n * 32 + nt * 16 + nlane;
                float mk = mask[b * NPH + t0 + nn];
#pragma unroll
                for (int v = 0; v < 8; ++v)
                    sy[nn * HH + m0 + v] = fmaxf((acc[mt][nt][v] + bias[m0 + v]) * mk, 0.f);
            }
        __syncthreads();
        // per-column stats over 192 contiguous channels; 4 partials per column
        int col = tid >> 2, part = tid & 3;
        {
            const float* p = &sy[col * HH + part * 48];
            float s1 = 0.f, s2 = 0.f;
            for (int m = 0; m < 48; ++m) { float v = p[m]; s1 += v; s2 += v * v; }
            sp1[col * 4 + part] = s1;
            sp2[col * 4 + part] = s2;
        }
        __syncthreads();
        if (part == 0) {
            float t1 = 0.f, t2 = 0.f;
            for (int j = 0; j < 4; ++j) { t1 += sp1[col * 4 + j]; t2 += sp2[col * 4 + j]; }
            float mean = t1 * (1.f / 192.f);
            float var  = t2 * (1.f / 192.f) - mean * mean;
            smean[col] = mean;
            srstd[col] = rsqrtf(var + 1e-5f);
        }
        __syncthreads();
        // normalize, scale, mask, store as 16B chunks: 64 cols x 24 groups of 8 channels
        for (int u = tid; u < 64 * 24; u += 256) {
            int nn = u / 24, m0 = (u % 24) * 8;
            float mean = smean[nn], rstd = srstd[nn];
            float mk = mask[b * NPH + t0 + nn];
            v8bf o;
#pragma unroll
            for (int j = 0; j < 8; ++j) {
                float v = (sy[nn * HH + m0 + j] - mean) * rstd * lng[m0 + j] + lnb[m0 + j];
                o[j] = (__bf16)(v * mk);
            }
            *(v8bf*)&dst[(size_t)(t0 + nn + 1) * HH + m0] = o;
        }
    }
}

// ---------------- final 4x192 projection (t-major input, contiguous dot) ----------------
__global__ void final_kernel(const __bf16* __restrict__ h, const float* __restrict__ lw,
                             const float* __restrict__ lb, const float* __restrict__ mask,
                             float* __restrict__ out) {
    int flat = blockIdx.x * blockDim.x + threadIdx.x;  // [b][o][t]
    if (flat >= NB * NOUT * NPH) return;
    int t = flat % NPH;
    int o = (flat / NPH) % NOUT;
    int b = flat / (NPH * NOUT);
    const __bf16* hb = h + (size_t)b * TPAD * HH + (size_t)(t + 1) * HH;
    const float*  wo = lw + o * HH;
    float acc = lb[o];
#pragma unroll
    for (int m = 0; m < HH; m += 8) {
        v8bf hv = *(const v8bf*)&hb[m];
#pragma unroll
        for (int j = 0; j < 8; ++j) acc += (float)hv[j] * wo[m + j];
    }
    out[flat] = acc * mask[b * NPH + t];
}

extern "C" void kernel_launch(void* const* d_in, const int* in_sizes, int n_in,
                              void* d_out, int out_size, void* d_ws, size_t ws_size,
                              hipStream_t stream) {
    const float* x      = (const float*)d_in[0];
    const float* xmask  = (const float*)d_in[1];
    const int*   w      = (const int*)  d_in[2];
    const float* pre_w  = (const float*)d_in[3];
    const float* pre_b  = (const float*)d_in[4];
    const float* c0w    = (const float*)d_in[5];
    const float* c0b    = (const float*)d_in[6];
    const float* ln0g   = (const float*)d_in[7];
    const float* ln0b   = (const float*)d_in[8];
    const float* c1w    = (const float*)d_in[9];
    const float* c1b    = (const float*)d_in[10];
    const float* ln1g   = (const float*)d_in[11];
    const float* ln1b   = (const float*)d_in[12];
    const float* lin_w  = (const float*)d_in[13];
    const float* lin_b  = (const float*)d_in[14];
    float* out = (float*)d_out;

    char* ws = (char*)d_ws;
    size_t off = 0;
    auto alloc = [&](size_t bytes) -> void* {
        void* p = ws + off;
        off += (bytes + 255) & ~(size_t)255;
        return p;
    };
    int*    ends  = (int*)   alloc((size_t)NB * NPH * 4);
    __bf16* preWb = (__bf16*)alloc((size_t)HH * CIN * 2);
    __bf16* c0Wb  = (__bf16*)alloc((size_t)3 * HH * HH * 2);
    __bf16* c1Wb  = (__bf16*)alloc((size_t)3 * HH * HH * 2);
    __bf16* spec  = (__bf16*)alloc((size_t)NB * TPAD * HH * 2);
    __bf16* bufA  = (__bf16*)alloc((size_t)NB * TPAD * HH * 2);
    __bf16* bufB  = (__bf16*)alloc((size_t)NB * TPAD * HH * 2);

    scan_kernel<<<NB, NPH, 0, stream>>>(w, ends);
    cvt_mat_kernel<<<(HH * CIN + 255) / 256, 256, 0, stream>>>(pre_w, preWb, HH * CIN);
    cvt_conv_kernel<<<(HH * HH * 3 + 255) / 256, 256, 0, stream>>>(c0w, c0Wb);
    cvt_conv_kernel<<<(HH * HH * 3 + 255) / 256, 256, 0, stream>>>(c1w, c1Wb);
    zero_halo_kernel<<<(NB * 2 * HH + 255) / 256, 256, 0, stream>>>(spec, bufA, bufB);
    segmean_kernel<<<NB * CIN, 256, 0, stream>>>(x, ends, spec);

    dim3 grid(NPH / 64, NB);
    block_kernel<1, false><<<grid, 256, 0, stream>>>(preWb, pre_b, spec, xmask,
                                                     nullptr, nullptr, bufA);
    block_kernel<3, true><<<grid, 256, 0, stream>>>(c0Wb, c0b, bufA, xmask,
                                                    ln0g, ln0b, bufB);
    block_kernel<3, true><<<grid, 256, 0, stream>>>(c1Wb, c1b, bufB, xmask,
                                                    ln1g, ln1b, bufA);

    final_kernel<<<(NB * NOUT * NPH + 255) / 256, 256, 0, stream>>>(bufA, lin_w, lin_b,
                                                                    xmask, out);
}
